// HAN_26242250178590
// MI455X (gfx1250) — matrix operand order repaired and checked by reading, not compiled, not run on hardware
//
#include <hip/hip_runtime.h>
#include <hip/hip_bf16.h>

// ---------------------------------------------------------------------------
// HAN-style GNN forward for MI455X (gfx1250, wave32, WMMA).
//
// Heavy math = 51x [N,128]@[128,128] GEMMs -> v_wmma_f32_16x16x32_bf16 with
// fp32 accumulation (bias / mean-normalization / accumulate / activation all
// fused, resolved at compile time via templates). Segment-mean aggregation =
// relation-sliced fp32 atomic scatter; the [N,128] accumulator (<=102MB) fits
// in the 192MB L2, keeping atomics on-chip. All row counts are multiples of
// 16, so WMMA waves always run with full EXEC (uniform wave-level tail guard).
// tanh activation uses the CDNA5 hardware V_TANH_F32 transcendental.
// ---------------------------------------------------------------------------

#define NE      200000
#define NSUB    100000
#define NREG    50000
#define NR      8
#define NRS     4
#define NMP     3
#define E_EDGES 1000000
#define G_EDGES 500000

typedef __attribute__((ext_vector_type(16))) __bf16 bf16x16;
typedef __attribute__((ext_vector_type(8)))  float  f32x8;

__device__ __forceinline__ float fast_tanh(float x) {
#if __has_builtin(__builtin_amdgcn_tanhf)
  return __builtin_amdgcn_tanhf(x);  // single V_TANH_F32 TRANS op on gfx1250
#else
  return tanhf(x);
#endif
}

// ---------------------------- utility kernels ------------------------------

__global__ void fill0_kernel(float* __restrict__ p, size_t n) {
  size_t i = (size_t)blockIdx.x * blockDim.x + threadIdx.x;
  size_t s = (size_t)gridDim.x * blockDim.x;
  for (; i < n; i += s) p[i] = 0.0f;
}

// out[i,:] = in[idx[i],:]. One 32-lane group per row; row id is uniform per
// group so idx[] is fetched via the scalar path.
__global__ __launch_bounds__(256) void gather_rows_kernel(
    float* __restrict__ out, const float* __restrict__ in,
    const int* __restrict__ idx, int n) {
  long r = (long)blockIdx.x * 8 + (threadIdx.x >> 5);
  int  c = threadIdx.x & 31;
  if (r >= n) return;
  long s = idx[r];
  *(float4*)(out + r * 128 + c * 4) = *(const float4*)(in + s * 128 + c * 4);
}

// emb[n][mp][d] = feat[n][d]  for n < NREG
__global__ __launch_bounds__(256) void copy_emb_kernel(
    float* __restrict__ emb, const float* __restrict__ feat, int mp, int n) {
  long r = (long)blockIdx.x * 8 + (threadIdx.x >> 5);
  int  c = threadIdx.x & 31;
  if (r >= n) return;
  *(float4*)(emb + (r * NMP + mp) * 128 + c * 4) =
      *(const float4*)(feat + r * 128 + c * 4);
}

// -------------------- relation-sliced segment-sum scatter ------------------
// For edges with et[e]==rel: acc[dst] += x[src]; cnt[dst] += 1.
// 32 lanes per edge, 4 dims per lane; e is uniform per group -> src/dst/et
// go through the scalar path, only features use vector memory. acc fits in
// L2 -> fp32 atomics stay on-chip.
__global__ __launch_bounds__(256) void scatter_rel_kernel(
    const float* __restrict__ X, const int* __restrict__ src,
    const int* __restrict__ dst, const int* __restrict__ et, int rel,
    float* __restrict__ acc, float* __restrict__ cnt, int nedges) {
  long e = (long)blockIdx.x * 8 + (threadIdx.x >> 5);
  int  c = threadIdx.x & 31;
  if (e >= nedges) return;
  if (et[e] != rel) return;
  long s = src[e];
  long d = dst[e];
  float4 xv = *(const float4*)(X + s * 128 + c * 4);
  float* a = acc + d * 128 + c * 4;
  unsafeAtomicAdd(a + 0, xv.x);
  unsafeAtomicAdd(a + 1, xv.y);
  unsafeAtomicAdd(a + 2, xv.z);
  unsafeAtomicAdd(a + 3, xv.w);
  if (c == 0) unsafeAtomicAdd(cnt + d, 1.0f);
}

// ----------------------------- WMMA GEMM -----------------------------------
// OUT[N,128] (+)= act( (X * rowscale) @ W[128,128] + bias )
//   SCALED: rowscale = 1/max(cnt[row],1)
//   ACCUM:  init C with existing OUT
//   ACT:    0=none 1=tanh 2=relu (applied at final store)
// Requires N % 16 == 0 (holds for all call sites): a wave's 16-row strip is
// either fully active or fully inactive -> uniform tail guard, full EXEC.
// One block = 256 threads = 8 waves; each wave owns a 16-row strip and sweeps
// the 8 column tiles with 4 chained v_wmma_f32_16x16x32_bf16 per tile.
// W is converted to bf16 pre-swizzled into a B-fragment layout in LDS (k = 16*half + element).
template <int ACT, bool ACCUM, bool SCALED>
__global__ __launch_bounds__(256) void gemm128_kernel(
    const float* __restrict__ X, const float* __restrict__ Wt,
    const float* __restrict__ bias, const float* __restrict__ cnt,
    float* __restrict__ OUT, int N) {
  // [kt][nt][lane][v*2+h] : lanes 0-15 K=0..15, lanes 16-31 K=16..31 (per kt)
  __shared__ __bf16 sB[4][8][32][16];  // 32 KB

  const int tid = threadIdx.x;
  for (int e = tid; e < 128 * 128; e += 256) {
    int k = e >> 7, n = e & 127;
    int kt = k >> 5, kk = k & 31;
    int lane = (n & 15) | ((kk >> 4) << 4);
    int v = (kk & 15) >> 1, hh = kk & 1;
    sB[kt][n >> 4][lane][v * 2 + hh] = (__bf16)Wt[e];
  }
  __syncthreads();

  const int wave = tid >> 5;
  const int lane = tid & 31;
  const int col  = lane & 15;
  const int hi   = lane >> 4;
  const long rowBase = (long)blockIdx.x * 128 + wave * 16;
  if (rowBase >= N) return;  // uniform per wave (N % 16 == 0)

  const long myRow = rowBase + col;  // A-fragment row for this lane
  float scale = 1.0f;
  if (SCALED) scale = 1.0f / fmaxf(cnt[myRow], 1.0f);

  // Build A fragments (16x32, contiguous k per lane half, same order as sB):
  //   VGPR v holds K = hi*16 + v*2  and +1
  bf16x16 afrag[4];
  const float* xrow = X + myRow * 128;
#pragma unroll
  for (int kt = 0; kt < 4; ++kt) {
#pragma unroll
    for (int v = 0; v < 8; ++v) {
      int kbase = kt * 32 + hi * 16 + v * 2;
      float2 xv = *(const float2*)(xrow + kbase);
      afrag[kt][v * 2]     = (__bf16)(xv.x * scale);
      afrag[kt][v * 2 + 1] = (__bf16)(xv.y * scale);
    }
  }

  for (int nt = 0; nt < 8; ++nt) {
    // C layout: VGPR j -> row rowBase + j (+8 for lanes 16-31), col per lane.
    float* orow = OUT + (rowBase + hi * 8) * 128 + nt * 16 + col;
    float bval = bias ? bias[nt * 16 + col] : 0.0f;
    f32x8 c;
#pragma unroll
    for (int j = 0; j < 8; ++j)
      c[j] = (ACCUM ? orow[j * 128] : 0.0f) + bval;

    // Preload all 4 B fragments so the ds_loads overlap, then chain WMMAs.
    bf16x16 bf[4];
#pragma unroll
    for (int kt = 0; kt < 4; ++kt) {
      const __bf16* bp = &sB[kt][nt][lane][0];
#pragma unroll
      for (int x = 0; x < 16; ++x) bf[kt][x] = bp[x];
    }
#pragma unroll
    for (int kt = 0; kt < 4; ++kt)
      c = __builtin_amdgcn_wmma_f32_16x16x32_bf16(
          false, afrag[kt], false, bf[kt], (short)0, c, false, false);

#pragma unroll
    for (int j = 0; j < 8; ++j) {
      float v = c[j];
      if (ACT == 1) v = fast_tanh(v);
      if (ACT == 2) v = fmaxf(v, 0.0f);
      orow[j * 128] = v;
    }
  }
}

// --------------------------- attention kernels -----------------------------

// q[3,128] = metapath_emb[3,64] @ qp_W[64,128] + qp_b
__global__ void qproj_kernel(const float* __restrict__ me,
                             const float* __restrict__ Wq,
                             const float* __restrict__ bq,
                             float* __restrict__ q) {
  int t = threadIdx.x;
  if (t >= NMP * 128) return;
  int m = t >> 7, d = t & 127;
  float s = bq[d];
  for (int k = 0; k < 64; ++k) s += me[m * 64 + k] * Wq[k * 128 + d];
  q[t] = s;
}

// scores -> softmax over metapaths -> weighted sum: one wave per node.
__global__ __launch_bounds__(256) void attn_kernel(
    const float* __restrict__ emb, const float* __restrict__ q,
    float* __restrict__ h, int n) {
  long node = (long)blockIdx.x * 8 + (threadIdx.x >> 5);
  int  lane = threadIdx.x & 31;
  if (node >= n) return;
  const float* e0 = emb + node * (NMP * 128);
  float4 ev[NMP];
  float dots[NMP];
#pragma unroll
  for (int m = 0; m < NMP; ++m) {
    float4 e = *(const float4*)(e0 + m * 128 + lane * 4);
    float4 qv = *(const float4*)(q + m * 128 + lane * 4);
    float dd = e.x * qv.x + e.y * qv.y + e.z * qv.z + e.w * qv.w;
    for (int off = 16; off; off >>= 1) dd += __shfl_down(dd, off, 32);
    dots[m] = __shfl(dd, 0, 32);
    ev[m] = e;
  }
  const float invSC = 1.0f / 11.313708498984761f;  // 1/sqrt(128)
  float d0 = dots[0] * invSC, d1 = dots[1] * invSC, d2 = dots[2] * invSC;
  float mx = fmaxf(d0, fmaxf(d1, d2));
  float w0 = __expf(d0 - mx), w1 = __expf(d1 - mx), w2 = __expf(d2 - mx);
  float inv = 1.0f / (w0 + w1 + w2);
  w0 *= inv; w1 *= inv; w2 *= inv;
  float4 hv;
  hv.x = w0 * ev[0].x + w1 * ev[1].x + w2 * ev[2].x;
  hv.y = w0 * ev[0].y + w1 * ev[1].y + w2 * ev[2].y;
  hv.z = w0 * ev[0].z + w1 * ev[1].z + w2 * ev[2].z;
  hv.w = w0 * ev[0].w + w1 * ev[1].w + w2 * ev[2].w;
  *(float4*)(h + node * 128 + lane * 4) = hv;
}

// wsa[slot] += sum_{n,j} t[n,j] * W2[j]   (t already tanh'ed)
__global__ __launch_bounds__(256) void sa_reduce_kernel(
    const float* __restrict__ t, const float* __restrict__ W2,
    float* __restrict__ wsa, int slot, int n) {
  long total = (long)n * 128;
  float s = 0.0f;
  for (long i = (long)blockIdx.x * blockDim.x + threadIdx.x; i < total;
       i += (long)gridDim.x * blockDim.x)
    s += t[i] * W2[i & 127];
  __shared__ float red[256];
  red[threadIdx.x] = s;
  __syncthreads();
  for (int st = 128; st; st >>= 1) {
    if (threadIdx.x < st) red[threadIdx.x] += red[threadIdx.x + st];
    __syncthreads();
  }
  if (threadIdx.x == 0) unsafeAtomicAdd(&wsa[slot], red[0]);
}

__global__ void beta_kernel(const float* __restrict__ wsa,
                            float* __restrict__ beta, float invN) {
  if (threadIdx.x == 0) {
    float a = wsa[0] * invN, b = wsa[1] * invN;
    float m = fmaxf(a, b);
    float ea = __expf(a - m), eb = __expf(b - m);
    float inv = 1.0f / (ea + eb);
    beta[0] = ea * inv;
    beta[1] = eb * inv;
  }
}

// out[n] = [beta0*Ereg + beta1*task , E[:NREG]] . er_W + er_b
__global__ __launch_bounds__(256) void final_kernel(
    const float* __restrict__ Ereg, const float* __restrict__ task,
    const float* __restrict__ E, const float* __restrict__ erW,
    const float* __restrict__ erb, const float* __restrict__ beta,
    float* __restrict__ out, int n) {
  long node = (long)blockIdx.x * 8 + (threadIdx.x >> 5);
  int  lane = threadIdx.x & 31;
  if (node >= n) return;
  float b0 = beta[0], b1 = beta[1];
  float s = 0.0f;
  {
    float4 er = *(const float4*)(Ereg + node * 128 + lane * 4);
    float4 tk = *(const float4*)(task + node * 128 + lane * 4);
    float4 w  = *(const float4*)(erW + lane * 4);
    s += (b0 * er.x + b1 * tk.x) * w.x + (b0 * er.y + b1 * tk.y) * w.y +
         (b0 * er.z + b1 * tk.z) * w.z + (b0 * er.w + b1 * tk.w) * w.w;
  }
  {
    float4 ev = *(const float4*)(E + node * 128 + lane * 4);
    float4 w  = *(const float4*)(erW + 128 + lane * 4);
    s += ev.x * w.x + ev.y * w.y + ev.z * w.z + ev.w * w.w;
  }
  for (int off = 16; off; off >>= 1) s += __shfl_down(s, off, 32);
  if (lane == 0) out[node] = s + erb[0];
}

// ------------------------------ host side ----------------------------------

static inline int fill_blocks(size_t n) {
  size_t b = (n + 255) / 256;
  return (int)(b > 2048 ? 2048 : (b < 1 ? 1 : b));
}

// act: 0 none, 1 tanh, 2 relu. accum implies rowscale (cnt) in our usage.
static void launch_gemm(const float* X, const float* W, const float* bias,
                        const float* cnt, float* OUT, int N, bool accum,
                        int act, hipStream_t s) {
  int blocks = (N + 127) / 128;
  if (!accum) {
    if (act == 1)
      gemm128_kernel<1, false, false><<<blocks, 256, 0, s>>>(X, W, bias, cnt,
                                                             OUT, N);
    else
      gemm128_kernel<0, false, false><<<blocks, 256, 0, s>>>(X, W, bias, cnt,
                                                             OUT, N);
  } else {
    if (act == 1)
      gemm128_kernel<1, true, true><<<blocks, 256, 0, s>>>(X, W, bias, cnt,
                                                           OUT, N);
    else if (act == 2)
      gemm128_kernel<2, true, true><<<blocks, 256, 0, s>>>(X, W, bias, cnt,
                                                           OUT, N);
    else
      gemm128_kernel<0, true, true><<<blocks, 256, 0, s>>>(X, W, bias, cnt,
                                                           OUT, N);
  }
}

// One RGCN layer: OUT = act( X@root + b + sum_r mean_r(X)@W_r )
static void run_rgcn_layer(const float* X, float* OUT, const float* W,
                           const float* b, const float* root, const int* src,
                           const int* dst, const int* et, int nedges, int nrel,
                           int nnodes, int act, float* acc, float* cnt,
                           hipStream_t stream) {
  launch_gemm(X, root, b, nullptr, OUT, nnodes, false, 0, stream);
  int sblocks = (nedges + 7) / 8;
  for (int r = 0; r < nrel; ++r) {
    fill0_kernel<<<fill_blocks((size_t)nnodes * 128), 256, 0, stream>>>(
        acc, (size_t)nnodes * 128);
    fill0_kernel<<<fill_blocks(nnodes), 256, 0, stream>>>(cnt, (size_t)nnodes);
    scatter_rel_kernel<<<sblocks, 256, 0, stream>>>(X, src, dst, et, r, acc,
                                                    cnt, nedges);
    launch_gemm(acc, W + (size_t)r * 128 * 128, nullptr, cnt, OUT, nnodes,
                true, (r == nrel - 1) ? act : 0, stream);
  }
}

extern "C" void kernel_launch(void* const* d_in, const int* in_sizes, int n_in,
                              void* d_out, int out_size, void* d_ws,
                              size_t ws_size, hipStream_t stream) {
  (void)in_sizes; (void)n_in; (void)out_size; (void)ws_size;

  // ---- inputs (setup_inputs order) ----
  const int* edge_index = (const int*)d_in[0];
  const int* edge_type  = (const int*)d_in[1];
  const int* g_ei[NMP]   = {(const int*)d_in[2], (const int*)d_in[5],
                            (const int*)d_in[8]};
  const int* g_et[NMP]   = {(const int*)d_in[3], (const int*)d_in[6],
                            (const int*)d_in[9]};
  const int* g_eids[NMP] = {(const int*)d_in[4], (const int*)d_in[7],
                            (const int*)d_in[10]};
  const float* metapath_emb = (const float*)d_in[11];
  const float* task_emb     = (const float*)d_in[12];

  // ---- params (JAX dict pytree: keys sorted; lists in order; conv dict
  //      flattens as W, b, root) ----
  const float* E_emb = (const float*)d_in[13];
  const float* er_W  = (const float*)d_in[14];
  const float* er_b  = (const float*)d_in[15];
  const float *mpW[NMP][2], *mpb[NMP][2], *mproot[NMP][2];
  for (int i = 0; i < NMP; ++i)
    for (int l = 0; l < 2; ++l) {
      int base = 16 + (i * 2 + l) * 3;
      mpW[i][l]    = (const float*)d_in[base + 0];
      mpb[i][l]    = (const float*)d_in[base + 1];
      mproot[i][l] = (const float*)d_in[base + 2];
    }
  const float* pred_W = (const float*)d_in[34];
  const float* pred_b = (const float*)d_in[35];
  const float* qp_W   = (const float*)d_in[36];
  const float* qp_b   = (const float*)d_in[37];
  const float *rW[2], *rb[2], *rroot[2];
  for (int l = 0; l < 2; ++l) {
    int base = 38 + l * 3;
    rW[l]    = (const float*)d_in[base + 0];
    rb[l]    = (const float*)d_in[base + 1];
    rroot[l] = (const float*)d_in[base + 2];
  }
  const float* sa_W1 = (const float*)d_in[44];
  const float* sa_b1 = (const float*)d_in[45];
  const float* sa_W2 = (const float*)d_in[46];

  // ---- workspace layout (floats) ----
  float* Wp = (float*)d_ws;
  size_t o = 0;
  float* bufA = Wp + o; o += (size_t)NE * 128;    // layer ping; later feat A/B
  float* bufB = Wp + o; o += (size_t)NE * 128;    // E_final lives here
  float* acc  = Wp + o; o += (size_t)NE * 128;
  float* cnt  = Wp + o; o += (size_t)NE;
  float* emb  = Wp + o; o += (size_t)NREG * NMP * 128;
  float* q    = Wp + o; o += (size_t)NMP * 128;
  float* h    = Wp + o; o += (size_t)NREG * 128;  // also tanh scratch for SA
  float* Ereg = Wp + o; o += (size_t)NREG * 128;
  float* wsa  = Wp + o; o += 2;
  float* beta = Wp + o; o += 2;
  float* featA = bufA;                       // reuse after global RGCN
  float* featB = bufA + (size_t)NSUB * 128;

  // ---- global RGCN stack (tanh) ----
  run_rgcn_layer(E_emb, bufA, rW[0], rb[0], rroot[0], edge_index,
                 edge_index + E_EDGES, edge_type, E_EDGES, NR, NE, 1, acc, cnt,
                 stream);
  run_rgcn_layer(bufA, bufB, rW[1], rb[1], rroot[1], edge_index,
                 edge_index + E_EDGES, edge_type, E_EDGES, NR, NE, 1, acc, cnt,
                 stream);
  // E_final = bufB (bufA now free -> feat buffers)

  // ---- per-metapath towers (relu) ----
  int gatherBlocks = (NSUB + 7) / 8;
  int embBlocks    = (NREG + 7) / 8;
  for (int i = 0; i < NMP; ++i) {
    gather_rows_kernel<<<gatherBlocks, 256, 0, stream>>>(featA, bufB,
                                                         g_eids[i], NSUB);
    run_rgcn_layer(featA, featB, mpW[i][0], mpb[i][0], mproot[i][0], g_ei[i],
                   g_ei[i] + G_EDGES, g_et[i], G_EDGES, NRS, NSUB, 2, acc, cnt,
                   stream);
    run_rgcn_layer(featB, featA, mpW[i][1], mpb[i][1], mproot[i][1], g_ei[i],
                   g_ei[i] + G_EDGES, g_et[i], G_EDGES, NRS, NSUB, 2, acc, cnt,
                   stream);
    copy_emb_kernel<<<embBlocks, 256, 0, stream>>>(emb, featA, i, NREG);
  }

  // ---- metapath attention ----
  qproj_kernel<<<1, NMP * 128, 0, stream>>>(metapath_emb, qp_W, qp_b, q);
  attn_kernel<<<embBlocks, 256, 0, stream>>>(emb, q, h, NREG);
  launch_gemm(h, pred_W, pred_b, nullptr, Ereg, NREG, false, 0, stream);

  // ---- cross-task semantic attention ----
  fill0_kernel<<<1, 256, 0, stream>>>(wsa, 2);
  launch_gemm(Ereg, sa_W1, sa_b1, nullptr, h, NREG, false, 1, stream);
  sa_reduce_kernel<<<1024, 256, 0, stream>>>(h, sa_W2, wsa, 0, NREG);
  launch_gemm(task_emb, sa_W1, sa_b1, nullptr, h, NREG, false, 1, stream);
  sa_reduce_kernel<<<1024, 256, 0, stream>>>(h, sa_W2, wsa, 1, NREG);
  beta_kernel<<<1, 32, 0, stream>>>(wsa, beta, 1.0f / (float)NREG);

  // ---- final projection ----
  final_kernel<<<embBlocks, 256, 0, stream>>>(Ereg, task_emb, bufB, er_W, er_b,
                                              beta, (float*)d_out, NREG);
}
